// ScoreFPELoss_1675037245597
// MI455X (gfx1250) — matrix operand order repaired and checked by reading, not compiled, not run on hardware
//
#include <hip/hip_runtime.h>
#include <hip/hip_bf16.h>

typedef __attribute__((ext_vector_type(16))) _Float16 v16h;
typedef __attribute__((ext_vector_type(8)))  float    v8f;

#define WAVES 2

__device__ inline v8f wmma16(v16h a, v16h b, v8f c) {
  // D = A(16x32 f16) * B(32x16 f16) + C(16x16 f32)
  return __builtin_amdgcn_wmma_f32_16x16x32_f16(
      /*neg_a=*/false, a, /*neg_b=*/false, b,
      /*c_mod=*/(short)0, c, /*reuse_a=*/false, /*reuse_b=*/false);
}

// Branch-free tanh on the hardware transcendental unit:
// tanh(u) = 1 - 2/(exp(2u)+1);  exp(2u) = exp2(u * 2*log2(e)).
// Saturates correctly: u->-inf => e->0 => -1 ; u->+inf => e->inf => rcp->0 => +1.
__device__ inline float fast_tanh(float u) {
  float e = __builtin_amdgcn_exp2f(u * 2.8853900817779268f);
  return 1.0f - 2.0f * __builtin_amdgcn_rcpf(e + 1.0f);
}

// Pack A fragment (16xK tile slice, f32 source in LDS) per CDNA5 16-bit A layout:
//   lane holds row M = lane&15; half j -> K = kbase + (j<8?0:16) + (lane>=16?8:0) + (j&7)
__device__ inline v16h pack_A(const float* src, int lane, int ld, int kbase) {
  v16h a;
  int m = lane & 15;
  int khalf = (lane >> 4) * 8;
#pragma unroll
  for (int j = 0; j < 16; ++j) {
    int k = kbase + ((j < 8) ? 0 : 16) + khalf + (j & 7);
    a[j] = (_Float16)src[m * ld + k];
  }
  return a;
}

// ---------------------------------------------------------------------------
// Setup: pack weights into WMMA B-fragment layouts (f16) + constants c_k.
// B-frag element (lane, j): k_local = (j<8?0:16)+(lane>=16?8:0)+(j&7), n_local = lane&15.
// ---------------------------------------------------------------------------
__global__ void fpe_setup(const float* __restrict__ W1, const float* __restrict__ W2,
                          _Float16* __restrict__ W1B, _Float16* __restrict__ W2B,
                          _Float16* __restrict__ W2TB, _Float16* __restrict__ W1TB,
                          float* __restrict__ cvec) {
  int tid = threadIdx.x;
  // c_k = sum_i W1[i,k] * W2[k,i]   (batch-independent divergence weights)
  for (int k = tid; k < 128; k += 256) {
    float acc = 0.f;
    for (int i = 0; i < 32; ++i) acc += W1[i * 128 + k] * W2[k * 32 + i];
    cvec[k] = acc;
  }
  // Each packed set is 8 fragments * 512 halves = 4096 elements.
  for (int idx = tid; idx < 8 * 512; idx += 256) {
    int f    = idx >> 9;           // fragment id 0..7
    int rem  = idx & 511;
    int lane = rem >> 4;
    int j    = rem & 15;
    int kl = ((j < 8) ? 0 : 16) + ((lane >= 16) ? 8 : 0) + (j & 7);  // k within 32-slab
    int nl = lane & 15;                    // n within 16-tile
    int ks = f >> 1, nt2 = f & 1;          // for K=128 sets: f = ks*2 + nt
    // GEMM1 B: W1x  (K=32 input dim, N=128 hidden), fragment f = n-tile
    W1B[idx]  = (_Float16)W1[kl * 128 + f * 16 + nl];
    // GEMM2/3 B: W2  (K=128 hidden, N=32 out), fragment = ks*2+nt
    W2B[idx]  = (_Float16)W2[(ks * 32 + kl) * 32 + nt2 * 16 + nl];
    // GEMM4 B: W2^T (K=32 out-dim i, N=128 hidden), fragment f = n-tile
    W2TB[idx] = (_Float16)W2[(f * 16 + nl) * 32 + kl];
    // GEMM5 B: W1x^T (K=128 hidden, N=32 input dim), fragment = ks*2+nt
    W1TB[idx] = (_Float16)W1[(nt2 * 16 + nl) * 128 + (ks * 32 + kl)];
  }
}

// ---------------------------------------------------------------------------
// Fused FPE-loss kernel: one wave per 16-row batch tile, 40 WMMAs per tile.
// ---------------------------------------------------------------------------
__global__ __launch_bounds__(32 * WAVES) void fpe_main(
    const float* __restrict__ x, const float* __restrict__ t,
    const float* __restrict__ beta, const float* __restrict__ W1,
    const float* __restrict__ b1, const float* __restrict__ b2,
    const _Float16* __restrict__ W1B, const _Float16* __restrict__ W2B,
    const _Float16* __restrict__ W2TB, const _Float16* __restrict__ W1TB,
    const float* __restrict__ cvec, float* __restrict__ out) {
  __shared__ float XS[WAVES][16 * 32];    // x tile, later v = x + 2s
  __shared__ float HS[WAVES][16 * 128];   // h tile, later q tile
  __shared__ float SS[WAVES][16 * 32];    // s tile
  __shared__ float RS[WAVES][16 * 32];    // ds_dt tile, later |resid|
  __shared__ float TS[WAVES][16];
  __shared__ float BTS[WAVES][16];

  const int lane = threadIdx.x & 31;
  const int wv   = threadIdx.x >> 5;
  const int row0 = (blockIdx.x * WAVES + wv) * 16;

  float* xs = XS[wv];
  float* hs = HS[wv];
  float* ss = SS[wv];
  float* rs = RS[wv];
  const float* w1t = W1 + 32 * 128;  // last row of W1 multiplies t

  // ---- load batch tile -----------------------------------------------------
  for (int i = lane; i < 512; i += 32) xs[i] = x[row0 * 32 + i];
  if (lane < 16) {
    TS[wv][lane]  = t[row0 + lane];
    BTS[wv][lane] = beta[row0 + lane];
  }
  __syncthreads();

  const int hi   = lane >> 4;   // C-frag row offset selector
  const int ncol = lane & 15;   // C-frag column

  // ---- GEMM1: u = x @ W1x  (+ t*w1t + b1), h = tanh(u) ---------------------
  {
    v16h ax = pack_A(xs, lane, 32, 0);
#pragma unroll
    for (int nt = 0; nt < 8; ++nt) {
      v16h b = *(const v16h*)(W1B + nt * 512 + lane * 16);
      v8f c = {};
      c = wmma16(ax, b, c);
      int k  = nt * 16 + ncol;
      float wk = w1t[k];
      float bk = b1[k];
#pragma unroll
      for (int r = 0; r < 8; ++r) {
        int mr = r + 8 * hi;
        float u = c[r] + TS[wv][mr] * wk + bk;
        hs[mr * 128 + k] = fast_tanh(u);
      }
    }
  }
  __syncthreads();

  // ---- GEMM2: s = h @ W2 + b2 ; GEMM3: ds_dt = (g*w1t) @ W2 ----------------
  {
    v16h ah[4], ap[4];
#pragma unroll
    for (int ks = 0; ks < 4; ++ks) {
      ah[ks] = pack_A(hs, lane, 128, ks * 32);
      v16h a;
      int m = lane & 15;
      int khalf = hi * 8;
#pragma unroll
      for (int j = 0; j < 16; ++j) {
        int k = ks * 32 + ((j < 8) ? 0 : 16) + khalf + (j & 7);
        float h = hs[m * 128 + k];
        a[j] = (_Float16)((1.0f - h * h) * w1t[k]);
      }
      ap[ks] = a;
    }
#pragma unroll
    for (int nt = 0; nt < 2; ++nt) {
      v8f cs = {};
      v8f cd = {};
#pragma unroll
      for (int ks = 0; ks < 4; ++ks) {
        v16h b = *(const v16h*)(W2B + (ks * 2 + nt) * 512 + lane * 16);
        cs = wmma16(ah[ks], b, cs);
        cd = wmma16(ap[ks], b, cd);
      }
      int j = nt * 16 + ncol;
      float b2j = b2[j];
#pragma unroll
      for (int r = 0; r < 8; ++r) {
        int mr = r + 8 * hi;
        ss[mr * 32 + j] = cs[r] + b2j;
        rs[mr * 32 + j] = cd[r];
      }
    }
  }
  __syncthreads();

  // ---- v = x + 2s ----------------------------------------------------------
  for (int i = lane; i < 512; i += 32) xs[i] += 2.0f * ss[i];
  __syncthreads();

  // ---- GEMM4: a = v @ W2^T ; q = g*(a - 2*c*h)  (overwrites h tile) --------
  {
    v16h av = pack_A(xs, lane, 32, 0);
#pragma unroll
    for (int nt = 0; nt < 8; ++nt) {
      v16h b = *(const v16h*)(W2TB + nt * 512 + lane * 16);
      v8f c = {};
      c = wmma16(av, b, c);
      int k = nt * 16 + ncol;
      float ck2 = 2.0f * cvec[k];
#pragma unroll
      for (int r = 0; r < 8; ++r) {
        int mr = r + 8 * hi;
        float h = hs[mr * 128 + k];
        float g = 1.0f - h * h;
        hs[mr * 128 + k] = g * (c[r] - ck2 * h);
      }
    }
  }
  __syncthreads();

  // ---- GEMM5: gx = q @ W1x^T ; resid = ds_dt - 0.5*beta*(s + gx) -----------
  {
    v16h aq[4];
#pragma unroll
    for (int ks = 0; ks < 4; ++ks) aq[ks] = pack_A(hs, lane, 128, ks * 32);
#pragma unroll
    for (int nt = 0; nt < 2; ++nt) {
      v8f c = {};
#pragma unroll
      for (int ks = 0; ks < 4; ++ks) {
        v16h b = *(const v16h*)(W1TB + (ks * 2 + nt) * 512 + lane * 16);
        c = wmma16(aq[ks], b, c);
      }
      int j = nt * 16 + ncol;
#pragma unroll
      for (int r = 0; r < 8; ++r) {
        int mr = r + 8 * hi;
        float rv = rs[mr * 32 + j] -
                   0.5f * BTS[wv][mr] * (ss[mr * 32 + j] + c[r]);
        rs[mr * 32 + j] = fabsf(rv);
      }
    }
  }
  __syncthreads();

  // ---- loss = mean_j |resid| ----------------------------------------------
  if (lane < 16) {
    float acc = 0.f;
#pragma unroll
    for (int j = 0; j < 32; ++j) acc += rs[lane * 32 + j];
    out[row0 + lane] = acc * (1.0f / 32.0f);
  }
}

// ---------------------------------------------------------------------------
extern "C" void kernel_launch(void* const* d_in, const int* in_sizes, int n_in,
                              void* d_out, int out_size, void* d_ws, size_t ws_size,
                              hipStream_t stream) {
  const float* x    = (const float*)d_in[0];  // [B,32]
  const float* t    = (const float*)d_in[1];  // [B,1]
  const float* beta = (const float*)d_in[2];  // [B,1]
  const float* W1   = (const float*)d_in[3];  // [33,128]
  const float* b1   = (const float*)d_in[4];  // [128]
  const float* W2   = (const float*)d_in[5];  // [128,32]
  const float* b2   = (const float*)d_in[6];  // [32]
  float* out = (float*)d_out;

  _Float16* W1B  = (_Float16*)d_ws;
  _Float16* W2B  = W1B + 4096;
  _Float16* W2TB = W2B + 4096;
  _Float16* W1TB = W2TB + 4096;
  float*    cvec = (float*)(W1TB + 4096);

  const int B = in_sizes[1];  // t has B elements
  fpe_setup<<<1, 256, 0, stream>>>(W1, W2, W1B, W2B, W2TB, W1TB, cvec);

  const int tiles  = B / 16;
  const int blocks = tiles / WAVES;
  fpe_main<<<blocks, 32 * WAVES, 0, stream>>>(x, t, beta, W1, b1, b2,
                                              W1B, W2B, W2TB, W1TB, cvec, out);
}
